// SelfMultiHeadAttn_3143916060830
// MI455X (gfx1250) — matrix operand order repaired and checked by reading, not compiled, not run on hardware
//
#include <hip/hip_runtime.h>
#include <hip/hip_bf16.h>
#include <hip/hip_fp16.h>

#define HIDDEN 1024
#define HEADS 16
#define HEAD_D 64
#define BATCH 64
#define MAX_S 2048
#define LN_EPS 1e-5f

typedef __attribute__((ext_vector_type(16))) _Float16 v16h;
typedef __attribute__((ext_vector_type(8)))  float    v8f;

// ---------------------------------------------------------------------------
// CDNA5 async helpers: ASYNCcnt-tracked global->LDS copy (ISA 10.x / 15.18.3).
// Generic LDS pointers carry the LDS byte address in their low 32 bits
// (flat-aperture rule).
// ---------------------------------------------------------------------------
__device__ __forceinline__ void async_copy_f4_to_lds(const float* gp, float* lp) {
    asm volatile("global_load_async_to_lds_b128 %0, %1, off"
                 :: "v"((unsigned int)(unsigned long long)(uintptr_t)lp),
                    "v"((unsigned long long)(uintptr_t)gp)
                 : "memory");
}

template <int N>
__device__ __forceinline__ void wait_asynccnt() {
    asm volatile("s_wait_asynccnt %0" :: "i"(N) : "memory");
}

// ---------------------------------------------------------------------------
// Skinny GEMM: C[64, N] = A[64, K] @ W[K, N] + bias[N]
// 4 waves per block, one 16x16 M-tile each (full M=64 strip per block), so W
// is streamed exactly once. The 32x16 W tile is cooperatively staged into a
// double-buffered LDS tile with b128 loads (1 float4/thread/step); B-fragments
// are then built from LDS. f16 fragments, f32 accumulation via WMMA
// (layouts per CDNA5 ISA 7.12.2: 16-bit A 16x32, B 32x16, wave32).
// ---------------------------------------------------------------------------
__global__ void gemm64_wmma(const float* __restrict__ A,
                            const float* __restrict__ W,
                            const float* __restrict__ bias,
                            float* __restrict__ C,
                            int K, int N) {
    const int tid  = threadIdx.x;       // 128 threads = 4 waves
    const int wave = tid >> 5;          // m-tile index
    const int lane = tid & 31;
    const int half = lane >> 4;         // 0: lanes 0-15, 1: lanes 16-31
    const int l16  = lane & 15;
    const int n0   = blockIdx.x * 16;

    __shared__ float sW[2][32 * 16];    // double-buffered 32x16 W tile

    const int rr = tid >> 2;            // staging row   0..31
    const int c4 = (tid & 3) * 4;       // staging col4  {0,4,8,12}

    v8f acc = {};
    int p = 0;
    for (int kk = 0; kk < K; kk += 32, p ^= 1) {
        // cooperative stage of W[kk..kk+31, n0..n0+15] into sW[p]
        const float4 w4 = *(const float4*)(W + (size_t)(kk + rr) * N + n0 + c4);
        *(float4*)(&sW[p][rr * 16 + c4]) = w4;
        __syncthreads();   // tile visible to all 4 waves (ping-pong: 1 barrier/step)

        // B fragment: VGPR pair v holds rows 8*half+2v (+8 for v>=4), +1 at col l16
        v16h bfrag;
#pragma unroll
        for (int v = 0; v < 8; ++v) {
            const int kr = 8 * half + 2 * v + (v >= 4 ? 8 : 0);
            bfrag[2 * v]     = (_Float16)sW[p][kr * 16 + l16];
            bfrag[2 * v + 1] = (_Float16)sW[p][(kr + 1) * 16 + l16];
        }
        // A fragment: lane row = wave*16 + l16; K pairs 8*half+2j and 16+8*half+2j
        const float* Arow = A + (size_t)(wave * 16 + l16) * K + kk;
        v16h afrag;
#pragma unroll
        for (int j = 0; j < 4; ++j) {
            const int kOff = 8 * half + 2 * j;
            afrag[2 * j]         = (_Float16)Arow[kOff];
            afrag[2 * j + 1]     = (_Float16)Arow[kOff + 1];
            afrag[8 + 2 * j]     = (_Float16)Arow[16 + kOff];
            afrag[8 + 2 * j + 1] = (_Float16)Arow[16 + kOff + 1];
        }
        acc = __builtin_amdgcn_wmma_f32_16x16x32_f16(
            false, afrag, false, bfrag, (short)0, acc, false, false);
    }

    const float bv = bias[n0 + l16];
#pragma unroll
    for (int r = 0; r < 8; ++r) {
        const int m = wave * 16 + r + 8 * half;   // C/D layout: VGPR r -> M = r + 8*half
        C[(size_t)m * N + n0 + l16] = acc[r] + bv;
    }
}

// ---------------------------------------------------------------------------
// Extract q/k/v for the new token from the fused qkv row; L2-normalize k along
// head dim and round through fp16 (as the reference does). Workspace only —
// the input caches are never mutated.
// qkv row layout per head h: [h*192 .. h*192+64) q, +64 k, +128 v
// ---------------------------------------------------------------------------
__global__ void kv_prepare(const float* __restrict__ qkv,
                           float* __restrict__ q_ws,
                           float* __restrict__ k_new,
                           float* __restrict__ v_new) {
    const int bh = blockIdx.x;            // b*HEADS + h
    const int b  = bh >> 4, h = bh & 15;
    const int d  = threadIdx.x;           // 64 threads
    const float* base = qkv + (size_t)b * (3 * HIDDEN) + h * (3 * HEAD_D);

    const float qv = base[d];
    const float kv = base[HEAD_D + d];
    const float vv = base[2 * HEAD_D + d];

    __shared__ float red[64];
    red[d] = kv * kv;
    __syncthreads();
    for (int off = 32; off > 0; off >>= 1) {
        if (d < off) red[d] += red[d + off];
        __syncthreads();
    }
    const float inv_norm = rsqrtf(red[0]);
    const float kn = __half2float(__float2half(kv * inv_norm));

    q_ws[bh * HEAD_D + d]  = qv;
    k_new[bh * HEAD_D + d] = kn;
    v_new[bh * HEAD_D + d] = vv;
}

// ---------------------------------------------------------------------------
// Decode attention, one workgroup (256 threads) per (b,h).
// Pass 1: per-thread dot products stream K rows (b128 + global_prefetch_b8).
// Pass 2: V tiles (32 rows x 64 f32) are software-pipelined into LDS with
// global_load_async_to_lds_b128, double-buffered two tiles deep, synchronized
// with s_wait_asynccnt + workgroup barriers. Position s == L (the new token)
// comes from k_new/v_new; the input caches are never written.
// ---------------------------------------------------------------------------
__global__ void attn_decode(const float* __restrict__ k_cache,
                            const float* __restrict__ v_cache,
                            const int*   __restrict__ seqlens,
                            const float* __restrict__ q_ws,
                            const float* __restrict__ k_new,
                            const float* __restrict__ v_new,
                            float* __restrict__ attn_out) {
    const int bh  = blockIdx.x;
    const int b   = bh >> 4, h = bh & 15;
    const int tid = threadIdx.x;          // 256 threads
    const int L   = seqlens[b];           // positions 0..L valid (L+1 total)

    __shared__ float sQ[HEAD_D];
    __shared__ float sScore[MAX_S];
    __shared__ float sRed[256];
    __shared__ float sV[2][32 * HEAD_D];  // double-buffered V tile (32 rows)

    if (tid < HEAD_D) sQ[tid] = q_ws[bh * HEAD_D + tid];
    __syncthreads();

    // ---- pass 1: scores -------------------------------------------------
    const float* kc = k_cache + (size_t)b * MAX_S * HIDDEN + h * HEAD_D;
    float lmax = -INFINITY;
    for (int s = tid; s <= L; s += 256) {
        const float* kp = (s == L) ? (k_new + bh * HEAD_D)
                                   : (kc + (size_t)s * HIDDEN);
        if (s + 256 < L) __builtin_prefetch(kc + (size_t)(s + 256) * HIDDEN, 0, 0);
        float acc = 0.f;
#pragma unroll
        for (int d0 = 0; d0 < HEAD_D; d0 += 4) {
            const float4 kq = *(const float4*)(kp + d0);
            acc += kq.x * sQ[d0] + kq.y * sQ[d0 + 1] + kq.z * sQ[d0 + 2] + kq.w * sQ[d0 + 3];
        }
        sScore[s] = acc;
        lmax = fmaxf(lmax, acc);
    }

    sRed[tid] = lmax;
    __syncthreads();
    for (int off = 128; off > 0; off >>= 1) {
        if (tid < off) sRed[tid] = fmaxf(sRed[tid], sRed[tid + off]);
        __syncthreads();
    }
    const float gmax = sRed[0];
    __syncthreads();

    float lsum = 0.f;
    for (int s = tid; s <= L; s += 256) {
        const float p = __expf(sScore[s] - gmax);
        sScore[s] = p;
        lsum += p;
    }
    sRed[tid] = lsum;
    __syncthreads();
    for (int off = 128; off > 0; off >>= 1) {
        if (tid < off) sRed[tid] += sRed[tid + off];
        __syncthreads();
    }
    const float inv_sum = 1.f / sRed[0];
    __syncthreads();

    // ---- pass 2: sum p[s] * V[s], async-pipelined V tiles ---------------
    const int d = tid & 63;
    const int g = tid >> 6;               // 4 s-groups
    const float* vc = v_cache + (size_t)b * MAX_S * HIDDEN + h * HEAD_D;

    const int TS = 32;                    // tile rows
    const int nt = (L + TS - 1) / TS;     // tiles covering s in [0, L); OOB rows <= 2047 stay in-allocation

    // each thread issues 2 async b128 ops per tile (512 float4 slots / 256 threads)
    auto stage = [&](int t, int pbuf) {
        const float* gbase = vc + (size_t)t * TS * HIDDEN;
#pragma unroll
        for (int u = 0; u < 2; ++u) {
            const int f  = tid + u * 256;       // float4 slot
            const int r  = f >> 4;              // tile row 0..31
            const int cc = (f & 15) * 4;        // col 0..60
            async_copy_f4_to_lds(gbase + (size_t)r * HIDDEN + cc,
                                 &sV[pbuf][r * HEAD_D + cc]);
        }
    };

    float acc = 0.f;
    if (nt > 0) stage(0, 0);
    if (nt > 1) stage(1, 1);
    for (int t = 0; t < nt; ++t) {
        if (t + 1 < nt) wait_asynccnt<2>();     // tile t landed; tile t+1 may be in flight
        else            wait_asynccnt<0>();
        __syncthreads();                        // all waves' async data visible
        const int smax = L - t * TS;            // valid rows in this tile (clamped below)
        const int lim  = smax < TS ? smax : TS;
        const float* vt = sV[t & 1];
        for (int sl = g; sl < lim; sl += 4)
            acc += sScore[t * TS + sl] * vt[sl * HEAD_D + d];
        __syncthreads();                        // everyone done reading buf before re-stage
        if (t + 2 < nt) stage(t + 2, t & 1);
    }
    // new token at s == L (added once, by group 0)
    if (g == 0) acc += sScore[L] * v_new[bh * HEAD_D + d];

    sRed[tid] = acc;
    __syncthreads();
    if (tid < HEAD_D) {
        const float o = (sRed[tid] + sRed[tid + 64] + sRed[tid + 128] + sRed[tid + 192]) * inv_sum;
        attn_out[(size_t)b * HIDDEN + h * HEAD_D + tid] = o;  // heads concatenated = reshape(B, HIDDEN)
    }
}

// ---------------------------------------------------------------------------
// Residual + dual LayerNorm: seq_out = LN(seq + postx), cand_out = LN(candc + postx)
// One block per row; outputs concatenated in tuple order.
// ---------------------------------------------------------------------------
__global__ void ln_final(const float* __restrict__ seq,
                         const float* __restrict__ candc,
                         const float* __restrict__ postx,
                         const float* __restrict__ gamma,
                         const float* __restrict__ beta,
                         float* __restrict__ out) {
    const int row = blockIdx.x;           // 64
    const int tid = threadIdx.x;          // 256, each owns 4 strided elements
    __shared__ float r1[256], r2[256];

    float x1[4], x2[4];
    float s1 = 0.f, s2 = 0.f;
#pragma unroll
    for (int i = 0; i < 4; ++i) {
        const int idx = tid + i * 256;
        const float px = postx[(size_t)row * HIDDEN + idx];
        x1[i] = seq[(size_t)row * HIDDEN + idx] + px;
        x2[i] = candc[(size_t)row * HIDDEN + idx] + px;
        s1 += x1[i];
        s2 += x2[i];
    }
    r1[tid] = s1; r2[tid] = s2;
    __syncthreads();
    for (int off = 128; off > 0; off >>= 1) {
        if (tid < off) { r1[tid] += r1[tid + off]; r2[tid] += r2[tid + off]; }
        __syncthreads();
    }
    const float mu1 = r1[0] * (1.f / HIDDEN);
    const float mu2 = r2[0] * (1.f / HIDDEN);
    __syncthreads();

    float v1 = 0.f, v2 = 0.f;
#pragma unroll
    for (int i = 0; i < 4; ++i) {
        const float d1 = x1[i] - mu1, d2 = x2[i] - mu2;
        v1 += d1 * d1;
        v2 += d2 * d2;
    }
    r1[tid] = v1; r2[tid] = v2;
    __syncthreads();
    for (int off = 128; off > 0; off >>= 1) {
        if (tid < off) { r1[tid] += r1[tid + off]; r2[tid] += r2[tid + off]; }
        __syncthreads();
    }
    const float rs1 = rsqrtf(r1[0] * (1.f / HIDDEN) + LN_EPS);
    const float rs2 = rsqrtf(r2[0] * (1.f / HIDDEN) + LN_EPS);

#pragma unroll
    for (int i = 0; i < 4; ++i) {
        const int idx = tid + i * 256;
        const float gg = gamma[idx], bt = beta[idx];
        out[(size_t)row * HIDDEN + idx]                            = (x1[i] - mu1) * rs1 * gg + bt;
        out[(size_t)(BATCH * HIDDEN) + (size_t)row * HIDDEN + idx] = (x2[i] - mu2) * rs2 * gg + bt;
    }
}

extern "C" void kernel_launch(void* const* d_in, const int* in_sizes, int n_in,
                              void* d_out, int out_size, void* d_ws, size_t ws_size,
                              hipStream_t stream) {
    const float* seq      = (const float*)d_in[0];
    const float* cand_aa  = (const float*)d_in[1];
    const float* k_cache  = (const float*)d_in[2];
    const float* v_cache  = (const float*)d_in[3];
    const int*   seqlens  = (const int*)  d_in[4];
    const float* Wqkv     = (const float*)d_in[5];
    const float* bqkv     = (const float*)d_in[6];
    const float* Wc       = (const float*)d_in[7];
    const float* bc       = (const float*)d_in[8];
    const float* Wo       = (const float*)d_in[9];
    const float* bo       = (const float*)d_in[10];
    const float* gamma    = (const float*)d_in[11];
    const float* beta     = (const float*)d_in[12];
    float* out = (float*)d_out;

    float* ws    = (float*)d_ws;
    float* qkv   = ws;                                   // 64*3072
    float* candc = qkv   + (size_t)BATCH * 3 * HIDDEN;   // 64*1024
    float* q_ws  = candc + (size_t)BATCH * HIDDEN;       // 64*16*64
    float* k_new = q_ws  + (size_t)BATCH * HIDDEN;       // 64*16*64
    float* v_new = k_new + (size_t)BATCH * HIDDEN;       // 64*16*64
    float* attnO = v_new + (size_t)BATCH * HIDDEN;       // 64*1024
    float* postx = attnO + (size_t)BATCH * HIDDEN;       // 64*1024

    // 1) fused QKV projection + candidate projection (WMMA GEMMs, 4 waves/block)
    gemm64_wmma<<<(3 * HIDDEN) / 16, 128, 0, stream>>>(seq,     Wqkv, bqkv, qkv,   HIDDEN, 3 * HIDDEN);
    gemm64_wmma<<<HIDDEN / 16,       128, 0, stream>>>(cand_aa, Wc,   bc,   candc, HIDDEN, HIDDEN);

    // 2) new-token q/k/v (k L2-normalized + fp16-rounded), workspace only
    kv_prepare<<<BATCH * HEADS, HEAD_D, 0, stream>>>(qkv, q_ws, k_new, v_new);

    // 3) decode attention over cache + new token (async V pipeline)
    attn_decode<<<BATCH * HEADS, 256, 0, stream>>>(k_cache, v_cache, seqlens,
                                                   q_ws, k_new, v_new, attnO);

    // 4) output projection (WMMA GEMM)
    gemm64_wmma<<<HIDDEN / 16, 128, 0, stream>>>(attnO, Wo, bo, postx, HIDDEN, HIDDEN);

    // 5) residual + dual LayerNorm -> (seq_out, candidate_out)
    ln_final<<<BATCH, 256, 0, stream>>>(seq, candc, postx, gamma, beta, out);
}
